// VEncoderLayerPyG_55602646614357
// MI455X (gfx1250) — matrix operand order repaired and checked by reading, not compiled, not run on hardware
//
#include <hip/hip_runtime.h>
#include <hip/hip_bf16.h>

typedef __attribute__((ext_vector_type(16))) _Float16 v16h;
typedef __attribute__((ext_vector_type(8)))  float    v8f;

#define D_DIM 128
#define H_HEADS 8
#define C_CH 16
#define FF_DIM 512
#define EPS_LN 1e-5f
#define NEG_SLOPE 0.2f

// ---------- helpers ----------

__device__ __forceinline__ float wave_sum(float v) {
  #pragma unroll
  for (int m = 16; m >= 1; m >>= 1) v += __shfl_xor(v, m, 32);
  return v;
}

// order-preserving float -> uint key for atomicMax
__device__ __forceinline__ unsigned fkey(float f) {
  unsigned u = __float_as_uint(f);
  return (u & 0x80000000u) ? ~u : (u | 0x80000000u);
}
__device__ __forceinline__ float fkey_dec(unsigned k) {
  return (k & 0x80000000u) ? __uint_as_float(k ^ 0x80000000u)
                           : __uint_as_float(~k);
}

// A fragment (16x32 f16, ISA 7.12.2 layout) gathered from a row-major f32 tile
__device__ __forceinline__ v16h load_A_f32(const float* t, int lda, int k0, int lane) {
  const int m   = lane & 15;
  const int khi = (lane >> 4) << 3;   // 0 or 8
  v16h a = {};
  #pragma unroll
  for (int i = 0; i < 16; ++i) {
    int k = k0 + khi + i + ((i >> 3) << 3);  // i<8: +i ; i>=8: +i+8
    a[i] = (_Float16)t[m * lda + k];
  }
  return a;
}

// B fragment from fragment-major packed weights: 32B contiguous per lane
// layout: P[((kt*ntiles + nt)*32 + lane)*16 + i]
__device__ __forceinline__ v16h load_Bp(const _Float16* __restrict__ P, int ntiles,
                                        int kt, int nt, int lane) {
  return *(const v16h*)(P + (((kt * ntiles + nt) * 32 + lane) << 4));
}

#define WMMA_F16(A, B, C) \
  __builtin_amdgcn_wmma_f32_16x16x32_f16(false, (A), false, (B), (short)0, (C), false, false)

// ---------- utility kernels ----------

__global__ void k_zero(float* p, long long n) {
  long long i = (long long)blockIdx.x * blockDim.x + threadIdx.x;
  if (i < n) p[i] = 0.0f;
}

// pack f32 row-major W[K][Ncols] -> f16 B-fragment-major
__global__ void k_pack_B(const float* __restrict__ W, _Float16* __restrict__ P,
                         int K, int Ncols) {
  int idx = blockIdx.x * blockDim.x + threadIdx.x;   // one packed half per thread
  if (idx >= K * Ncols) return;
  int i      = idx & 15;
  int lane   = (idx >> 4) & 31;
  int frag   = idx >> 9;
  int ntiles = Ncols >> 4;
  int kt = frag / ntiles, nt = frag - kt * ntiles;
  int n = nt * 16 + (lane & 15);
  int k = kt * 32 + ((lane >> 4) << 3) + i + ((i >> 3) << 3);
  P[idx] = (_Float16)W[k * Ncols + n];
}

// ---------- kernel 1: h = x @ W_gat  (+ per-node attention logits) ----------
// one wave per 16-row tile; 8 N-tiles x 4 K-chunks = 32 WMMAs / wave
__global__ __launch_bounds__(128) void k_gat_linear(
    const float* __restrict__ x, const _Float16* __restrict__ Wg16,
    const float* __restrict__ att_src, const float* __restrict__ att_dst,
    float* __restrict__ h, float* __restrict__ a_s, float* __restrict__ a_d, int N)
{
  __shared__ float lds[4][16 * D_DIM];
  const int lane = threadIdx.x & 31;
  const int wv   = threadIdx.x >> 5;
  const int row0 = (blockIdx.x * 4 + wv) * 16;
  if (row0 >= N) return;
  float* xt = lds[wv];

  // stage 16x128 x-tile into LDS (float4, coalesced, uniform row guard)
  for (int it = 0; it < 16; ++it) {
    int gr = row0 + it;
    float4 v = make_float4(0.f, 0.f, 0.f, 0.f);
    if (gr < N) v = ((const float4*)(x + (long long)gr * D_DIM))[lane];
    ((float4*)(xt + it * D_DIM))[lane] = v;
  }

  v16h afr[4];
  #pragma unroll
  for (int kc = 0; kc < 4; ++kc) afr[kc] = load_A_f32(xt, D_DIM, kc * 32, lane);

  const int mh = (lane >> 4) << 3;
  const int ll = lane & 15;
  #pragma unroll
  for (int j = 0; j < 8; ++j) {
    v8f acc = (v8f){};
    #pragma unroll
    for (int kc = 0; kc < 4; ++kc)
      acc = WMMA_F16(afr[kc], load_Bp(Wg16, 8, kc, j, lane), acc);
    // overwrite LDS tile with h (the A-fragment regs already hold x)
    int n = j * 16 + ll;
    #pragma unroll
    for (int r = 0; r < 8; ++r) xt[(mh + r) * D_DIM + n] = acc[r];
  }

  // coalesced write-back of h from LDS (float4 rows, uniform guard)
  for (int it = 0; it < 16; ++it) {
    int gr = row0 + it;
    if (gr < N)
      ((float4*)(h + (long long)gr * D_DIM))[lane] =
          ((const float4*)(xt + it * D_DIM))[lane];
  }

  // a_s[n,head] = sum_c h[n,head,c]*att_src[head,c] ; likewise a_d
  #pragma unroll
  for (int i = 0; i < 4; ++i) {
    int p = lane * 4 + i;          // 0..127 -> (m, head)
    int m = p >> 3, head = p & 7, gr = row0 + m;
    if (gr < N) {
      float ss = 0.f, sd = 0.f;
      #pragma unroll
      for (int c = 0; c < C_CH; ++c) {
        float v = xt[m * D_DIM + head * C_CH + c];
        ss += v * att_src[head * C_CH + c];
        sd += v * att_dst[head * C_CH + c];
      }
      a_s[gr * H_HEADS + head] = ss;
      a_d[gr * H_HEADS + head] = sd;
    }
  }
}

// ---------- kernel 2: segment max over destinations ----------
__global__ void k_edge_max(const int* __restrict__ ei, int E, int N,
                           const float* __restrict__ a_s, const float* __restrict__ a_d,
                           unsigned* __restrict__ maxe)
{
  long long t = (long long)blockIdx.x * blockDim.x + threadIdx.x;
  long long total = (long long)(E + N) * H_HEADS;
  if (t >= total) return;
  int e = (int)(t >> 3), head = (int)(t & 7);
  int s, d;
  if (e < E) { s = ei[e]; d = ei[E + e]; } else { s = e - E; d = s; }
  float v = a_s[s * H_HEADS + head] + a_d[d * H_HEADS + head];
  v = (v >= 0.f) ? v : NEG_SLOPE * v;
  atomicMax(maxe + (long long)d * H_HEADS + head, fkey(v));
}

// ---------- kernel 3: fused exp / denom / weighted scatter-add ----------
// normalization deferred: agg = sum h[src]*alpha ; den = sum alpha
__global__ void k_edge_agg(const int* __restrict__ ei, int E, int N,
                           const float* __restrict__ a_s, const float* __restrict__ a_d,
                           const unsigned* __restrict__ maxe, const float* __restrict__ h,
                           float* __restrict__ den, float* __restrict__ agg)
{
  long long t = (long long)blockIdx.x * blockDim.x + threadIdx.x;
  long long total = (long long)(E + N) * 32;
  if (t >= total) return;
  int e = (int)(t >> 5), l = (int)(t & 31);
  int s, d;
  if (e < E) { s = ei[e]; d = ei[E + e]; } else { s = e - E; d = s; }
  int head = l >> 2;                     // 4 lanes (16 channels) per head
  float v = a_s[s * H_HEADS + head] + a_d[d * H_HEADS + head];
  v = (v >= 0.f) ? v : NEG_SLOPE * v;
  float m = fkey_dec(maxe[(long long)d * H_HEADS + head]);
  float alpha = __expf(v - m);
  if ((l & 3) == 0) atomicAdd(den + (long long)d * H_HEADS + head, alpha);
  float4 hv = ((const float4*)(h + (long long)s * D_DIM))[l];
  float* ap = agg + (long long)d * D_DIM + l * 4;
  atomicAdd(ap + 0, hv.x * alpha);
  atomicAdd(ap + 1, hv.y * alpha);
  atomicAdd(ap + 2, hv.z * alpha);
  atomicAdd(ap + 3, hv.w * alpha);
}

// ---------- kernel 4: fused  LN1 -> FF(GEMM,ReLU,GEMM) -> +res -> LN2 ----------
// one wave per 16-row tile; 128 + 128 = 256 WMMAs / wave
__global__ __launch_bounds__(64) void k_fused_ffn(
    const float* __restrict__ x, const float* __restrict__ agg,
    const float* __restrict__ den, const float* __restrict__ b_gat,
    const _Float16* __restrict__ W1p, const float* __restrict__ b1,
    const _Float16* __restrict__ W2p, const float* __restrict__ b2,
    const float* __restrict__ ln1g, const float* __restrict__ ln1b,
    const float* __restrict__ ln2g, const float* __restrict__ ln2b,
    float* __restrict__ out, int N)
{
  __shared__ float                  ldsA[2][16 * D_DIM];    // h1 (f32)
  __shared__ __align__(32) _Float16 ldsQ[2][16 * FF_DIM];   // hidden, A-fragment-packed f16
  const int lane = threadIdx.x & 31;
  const int wv   = threadIdx.x >> 5;
  const int row0 = (blockIdx.x * 2 + wv) * 16;
  if (row0 >= N) return;
  float*     At = ldsA[wv];
  _Float16*  Qp = ldsQ[wv];

  __builtin_prefetch(W1p, 0, 3);   // global_prefetch_b8 into L2
  __builtin_prefetch(W2p, 0, 3);

  // ---- phase 1: y = x + agg/den + b_gat ; h1 = LN1(y) -> LDS ----
  const int c0   = lane * 4;
  const int head = lane >> 2;
  for (int r = 0; r < 16; ++r) {
    int gr = row0 + r;
    float v0 = 0.f, v1 = 0.f, v2 = 0.f, v3 = 0.f;
    if (gr < N) {
      float4 xv = ((const float4*)(x   + (long long)gr * D_DIM))[lane];
      float4 av = ((const float4*)(agg + (long long)gr * D_DIM))[lane];
      float inv = 1.0f / den[gr * H_HEADS + head];
      v0 = xv.x + av.x * inv + b_gat[c0 + 0];
      v1 = xv.y + av.y * inv + b_gat[c0 + 1];
      v2 = xv.z + av.z * inv + b_gat[c0 + 2];
      v3 = xv.w + av.w * inv + b_gat[c0 + 3];
    }
    float mu = wave_sum(v0 + v1 + v2 + v3) * (1.0f / 128.0f);
    float d0 = v0 - mu, d1 = v1 - mu, d2 = v2 - mu, d3 = v3 - mu;
    float var = wave_sum(d0 * d0 + d1 * d1 + d2 * d2 + d3 * d3) * (1.0f / 128.0f);
    float rs = rsqrtf(var + EPS_LN);
    At[r * D_DIM + c0 + 0] = d0 * rs * ln1g[c0 + 0] + ln1b[c0 + 0];
    At[r * D_DIM + c0 + 1] = d1 * rs * ln1g[c0 + 1] + ln1b[c0 + 1];
    At[r * D_DIM + c0 + 2] = d2 * rs * ln1g[c0 + 2] + ln1b[c0 + 2];
    At[r * D_DIM + c0 + 3] = d3 * rs * ln1g[c0 + 3] + ln1b[c0 + 3];
  }

  // ---- phase 2: Q = relu(h1 @ W1 + b1), written A-fragment-packed into LDS ----
  v16h afr[4];
  #pragma unroll
  for (int kc = 0; kc < 4; ++kc) afr[kc] = load_A_f32(At, D_DIM, kc * 32, lane);

  const int mh = (lane >> 4) << 3;
  const int ll = lane & 15;
  // writer-side packed coords for GEMM2's A fragments (K2 = FF dim):
  //   chunk kc2 = j>>1 ; i = (ll&7) + 8*(j&1) ; lane' = m + 16*(ll>>3)
  const int lgrp = (ll >> 3) << 4;
  for (int j = 0; j < 32; ++j) {
    v8f acc = (v8f){};
    #pragma unroll
    for (int kc = 0; kc < 4; ++kc)
      acc = WMMA_F16(afr[kc], load_Bp(W1p, 32, kc, j, lane), acc);
    float bb = b1[j * 16 + ll];
    int ibase = ((j >> 1) * 32 + lgrp + mh) * 16 + (ll & 7) + ((j & 1) << 3);
    #pragma unroll
    for (int r = 0; r < 8; ++r) {
      float tv = acc[r] + bb;
      Qp[ibase + r * 16] = (_Float16)(tv > 0.f ? tv : 0.f);  // lane' = mh+r (+lgrp)
    }
  }

  // ---- phase 3: F = Q @ W2 + b2 ; z = h1 + F ----
  v8f acc2[8];
  #pragma unroll
  for (int j = 0; j < 8; ++j) acc2[j] = (v8f){};
  for (int kc = 0; kc < 16; ++kc) {
    v16h a = *(const v16h*)(Qp + ((kc * 32 + lane) << 4));   // 32B contiguous
    #pragma unroll
    for (int j = 0; j < 8; ++j)
      acc2[j] = WMMA_F16(a, load_Bp(W2p, 8, kc, j, lane), acc2[j]);
  }
  float* Zt = (float*)Qp;  // reuse hidden buffer (all Q reads complete, same wave)
  #pragma unroll
  for (int j = 0; j < 8; ++j) {
    int n = j * 16 + ll;
    float bb = b2[n];
    #pragma unroll
    for (int r = 0; r < 8; ++r) {
      int m = mh + r;
      Zt[m * D_DIM + n] = acc2[j][r] + bb + At[m * D_DIM + n];
    }
  }

  // ---- phase 4: out = LN2(z) ----
  for (int r = 0; r < 16; ++r) {
    int gr = row0 + r;
    float4 zv = ((const float4*)(Zt + r * D_DIM))[lane];
    float mu = wave_sum(zv.x + zv.y + zv.z + zv.w) * (1.0f / 128.0f);
    float d0 = zv.x - mu, d1 = zv.y - mu, d2 = zv.z - mu, d3 = zv.w - mu;
    float var = wave_sum(d0 * d0 + d1 * d1 + d2 * d2 + d3 * d3) * (1.0f / 128.0f);
    float rs = rsqrtf(var + EPS_LN);
    if (gr < N) {
      float* op = out + (long long)gr * D_DIM + c0;
      op[0] = d0 * rs * ln2g[c0 + 0] + ln2b[c0 + 0];
      op[1] = d1 * rs * ln2g[c0 + 1] + ln2b[c0 + 1];
      op[2] = d2 * rs * ln2g[c0 + 2] + ln2b[c0 + 2];
      op[3] = d3 * rs * ln2g[c0 + 3] + ln2b[c0 + 3];
    }
  }
}

// ---------- host launch ----------
extern "C" void kernel_launch(void* const* d_in, const int* in_sizes, int n_in,
                              void* d_out, int out_size, void* d_ws, size_t ws_size,
                              hipStream_t stream) {
  (void)n_in; (void)out_size; (void)ws_size;
  const float* x       = (const float*)d_in[0];
  const int*   ei      = (const int*)  d_in[1];
  const float* W_gat   = (const float*)d_in[2];
  const float* att_src = (const float*)d_in[3];
  const float* att_dst = (const float*)d_in[4];
  const float* b_gat   = (const float*)d_in[5];
  const float* W1      = (const float*)d_in[6];
  const float* b1      = (const float*)d_in[7];
  const float* W2      = (const float*)d_in[8];
  const float* b2      = (const float*)d_in[9];
  const float* ln1g    = (const float*)d_in[10];
  const float* ln1b    = (const float*)d_in[11];
  const float* ln2g    = (const float*)d_in[12];
  const float* ln2b    = (const float*)d_in[13];
  const int N = in_sizes[0] / D_DIM;
  const int E = in_sizes[1] / 2;
  float* out = (float*)d_out;

  // workspace layout (all offsets 32B-aligned)
  char* w = (char*)d_ws;
  float*    h     = (float*)w;     w += (size_t)N * D_DIM   * 4;
  float*    a_s   = (float*)w;     w += (size_t)N * H_HEADS * 4;
  float*    a_d   = (float*)w;     w += (size_t)N * H_HEADS * 4;
  float*    zreg  = (float*)w;                       // [maxe | den | agg] contiguous
  unsigned* maxe  = (unsigned*)w;  w += (size_t)N * H_HEADS * 4;
  float*    den   = (float*)w;     w += (size_t)N * H_HEADS * 4;
  float*    agg   = (float*)w;     w += (size_t)N * D_DIM   * 4;
  _Float16* Wg16  = (_Float16*)w;  w += (size_t)D_DIM * D_DIM  * 2;
  _Float16* W1p   = (_Float16*)w;  w += (size_t)D_DIM * FF_DIM * 2;
  _Float16* W2p   = (_Float16*)w;  w += (size_t)FF_DIM * D_DIM * 2;

  // weights -> fragment-major f16
  k_pack_B<<<(D_DIM * D_DIM  + 255) / 256, 256, 0, stream>>>(W_gat, Wg16, D_DIM, D_DIM);
  k_pack_B<<<(D_DIM * FF_DIM + 255) / 256, 256, 0, stream>>>(W1,    W1p,  D_DIM, FF_DIM);
  k_pack_B<<<(FF_DIM * D_DIM + 255) / 256, 256, 0, stream>>>(W2,    W2p,  FF_DIM, D_DIM);

  // zero maxe+den+agg (keys of 0 act as -inf)
  long long zn = (long long)N * (H_HEADS + H_HEADS + D_DIM);
  k_zero<<<(unsigned)((zn + 255) / 256), 256, 0, stream>>>(zreg, zn);

  // GAT projection + attention logits (WMMA)
  int tiles = (N + 15) / 16;
  k_gat_linear<<<(tiles + 3) / 4, 128, 0, stream>>>(x, Wg16, att_src, att_dst,
                                                    h, a_s, a_d, N);

  // edge phase
  long long tmax = (long long)(E + N) * H_HEADS;
  k_edge_max<<<(unsigned)((tmax + 255) / 256), 256, 0, stream>>>(ei, E, N, a_s, a_d, maxe);
  long long tagg = (long long)(E + N) * 32;
  k_edge_agg<<<(unsigned)((tagg + 255) / 256), 256, 0, stream>>>(ei, E, N, a_s, a_d,
                                                                 maxe, h, den, agg);

  // fused LN1 + FFN + LN2 (WMMA)
  k_fused_ffn<<<(tiles + 1) / 2, 64, 0, stream>>>(x, agg, den, b_gat, W1p, b1, W2p, b2,
                                                  ln1g, ln1b, ln2g, ln2b, out, N);
}
